// EnergyGatedDeltaModel_3204045603495
// MI455X (gfx1250) — compile-verified, matrix operand order checked by reading, not run on hardware
//
#include <hip/hip_runtime.h>

#define B_  256
#define L_  2048
#define H_  128
#define V_  32000

typedef __bf16 bf16;
typedef __attribute__((ext_vector_type(16))) __bf16        v16bf;
typedef __attribute__((ext_vector_type(8)))  float         v8f;
typedef __attribute__((ext_vector_type(8)))  unsigned int  v8u;
typedef __attribute__((ext_vector_type(4)))  unsigned int  v4ui;
typedef __attribute__((ext_vector_type(4)))  int           v4i;
typedef __attribute__((ext_vector_type(8)))  int           v8i;

#if __has_builtin(__builtin_amdgcn_tensor_load_to_lds) && \
    __has_builtin(__builtin_amdgcn_s_wait_tensorcnt)
#define USE_TDM_GATHER 1
#else
#define USE_TDM_GATHER 0
#endif

// --- WMMA fragment index helpers (16-bit A/B layout, ISA 7.12.2) -----------
// For lane l: half = l>>4. VGPR rr holds K pair starting at:
//   kbase(rr,half) = (rr<4 ? 2*rr : 16+2*(rr-4)) + 8*half
__device__ __forceinline__ int kbase(int rr, int kh) {
  return ((rr < 4) ? (2 * rr) : (16 + 2 * (rr - 4))) + 8 * kh;
}
// Same mapping expressed as a dword index into a 32-element (16-dword) K row.
__device__ __forceinline__ int afrag_u32(int rr, int kh) {
  return (rr & 3) + 4 * kh + ((rr >= 4) ? 8 : 0);
}

__device__ __forceinline__ v8f wmma_bf16(v16bf a, v16bf b, v8f c) {
  return __builtin_amdgcn_wmma_f32_16x16x32_bf16(false, a, false, b, (short)0, c,
                                                 false, false);
}

// --- Pack f32 row-major [K][N] weights into bf16 B-fragment-major layout ---
// dst[((kt*(N/16)+nt)*32 + lane)*16 + e] = (bf16)src[k][n]
__global__ __launch_bounds__(256) void pack_b(const float* __restrict__ src,
                                              bf16* __restrict__ dst,
                                              int K, int N) {
  int idx = blockIdx.x * 256 + threadIdx.x;
  if (idx >= K * N) return;
  int e = idx & 15, lane = (idx >> 4) & 31, tile = idx >> 9;
  int ntiles = N >> 4;
  int nt = tile % ntiles, kt = tile / ntiles;
  int rr = e >> 1, p = e & 1, kh = lane >> 4;
  int k = kt * 32 + kbase(rr, kh) + p;
  int n = nt * 16 + (lane & 15);
  dst[idx] = (bf16)src[(size_t)k * N + n];
}

// --- Encoder: TDM-gather + MLP (WMMA bf16) + residual + LayerNorm ----------
// 64 tokens per block, 256 threads (8 waves). GEMM1: 64x256x128, GEMM2:
// 64x128x256. Weights streamed from L2 (resident in 192MB L2).
__global__ __launch_bounds__(256) void encoder_kernel(
    const int* __restrict__ seq, const float* __restrict__ embed,
    const bf16* __restrict__ W1p, const float* __restrict__ b1,
    const bf16* __restrict__ W2p, const float* __restrict__ b2,
    const float* __restrict__ gamma, const float* __restrict__ beta,
    float* __restrict__ h_all) {
  __shared__ float sEmb[64 * 128];                                 // 32 KB f32
  __shared__ union { bf16 act[64 * 256]; float y[64 * 128]; } sU;  // 32 KB

  const int tid = threadIdx.x;
  const size_t tokbase = (size_t)blockIdx.x * 64;
  const int wave = tid >> 5, lane = tid & 31;

  // Phase 0: gather 64 embedding rows (f32) into LDS.
#if USE_TDM_GATHER
  // TDM gather mode: waves 0..3 each fetch 16 indexed rows of 128 f32.
  if (wave < 4) {
    const int tb = (int)tokbase + wave * 16;
    unsigned idx[16];
    #pragma unroll
    for (int j = 0; j < 16; ++j) idx[j] = (unsigned)seq[tb + j] & 0xFFFFu;

    unsigned lds_base = (unsigned)(size_t)(&sEmb[wave * 16 * 128]);
    unsigned long long ga = (unsigned long long)(size_t)embed;

    // D# group 0: count=1, gather_mode=1 (16-bit idx), lds_addr, global_addr,
    // type=2 ("image")
    v4ui g0 = { 1u | (1u << 31),
                lds_base,
                (unsigned)(ga & 0xFFFFFFFFu),
                (unsigned)((ga >> 32) & 0x1FFFFFFu) | (2u << 30) };
    // D# group 1: data_size=4B; tensor_dim0=128; tensor_dim1=32000 rows;
    // tile_dim0=128; tile_dim1=16 valid indices; dim0_stride=128.
    v8i g1 = { (int)(2u << 16),
               (int)(128u << 16),
               (int)((32000u & 0xFFFFu) << 16),
               (int)(128u << 16),
               16,
               128,
               0, 0 };
    // D# groups 2/3: 16-bit row indices, two per dword.
    v4i g2 = { (int)(idx[0] | (idx[1] << 16)),
               (int)(idx[2] | (idx[3] << 16)),
               (int)(idx[4] | (idx[5] << 16)),
               (int)(idx[6] | (idx[7] << 16)) };
    v4i g3 = { (int)(idx[8]  | (idx[9]  << 16)),
               (int)(idx[10] | (idx[11] << 16)),
               (int)(idx[12] | (idx[13] << 16)),
               (int)(idx[14] | (idx[15] << 16)) };
#if __clang_major__ >= 23
    __builtin_amdgcn_tensor_load_to_lds(g0, g1, g2, g3, (v8i)0, 0);
#else
    __builtin_amdgcn_tensor_load_to_lds(g0, g1, g2, g3, 0);
#endif
    __builtin_amdgcn_s_wait_tensorcnt(0);
  }
#else
  for (int i = tid; i < 64 * 128; i += 256) {
    int t = i >> 7, h = i & 127;
    int v = seq[tokbase + t];
    sEmb[i] = embed[(size_t)v * H_ + h];
  }
#endif
  __syncthreads();

  const int m = lane & 15, kh = lane >> 4;
  const int mt = wave & 3;
  const int row = mt * 16 + m;

  // Phase 1: GEMM1 (relu(e@W1+b1)) -> sU.act (bf16)
  {
    const int ntbase = (wave >> 2) * 8;
    for (int j = 0; j < 8; ++j) {
      int nt = ntbase + j;
      int col = nt * 16 + m;
      float bias = b1[col];
      v8f acc;
      #pragma unroll
      for (int e = 0; e < 8; ++e) acc[e] = bias;
      #pragma unroll
      for (int kk = 0; kk < 4; ++kk) {
        v16bf a;
        #pragma unroll
        for (int rr = 0; rr < 8; ++rr) {
          const float* p = sEmb + row * 128 + kk * 32 + kbase(rr, kh);
          a[2 * rr]     = (bf16)p[0];
          a[2 * rr + 1] = (bf16)p[1];
        }
        const v8u* bp =
            (const v8u*)(W1p + (((size_t)kk * 16 + nt) * 32 + lane) * 16);
        v16bf b = __builtin_bit_cast(v16bf, *bp);
        if (kk < 3)
          __builtin_prefetch(
              W1p + (((size_t)(kk + 1) * 16 + nt) * 32 + lane) * 16, 0, 0);
        acc = wmma_bf16(a, b, acc);
      }
      #pragma unroll
      for (int e = 0; e < 8; ++e) {
        float z = acc[e] > 0.f ? acc[e] : 0.f;
        sU.act[(mt * 16 + e + 8 * kh) * 256 + col] = (bf16)z;
      }
    }
  }
  __syncthreads();

  // Phase 2: GEMM2 (+ b2 + residual), accumulate in registers
  float yv[4][8];
  {
    const int ntbase2 = (wave >> 2) * 4;
    for (int j = 0; j < 4; ++j) {
      int nt = ntbase2 + j;
      int col = nt * 16 + m;
      float bias = b2[col];
      v8f acc;
      #pragma unroll
      for (int e = 0; e < 8; ++e) acc[e] = bias;
      #pragma unroll
      for (int kk = 0; kk < 8; ++kk) {
        const unsigned int* rowU =
            (const unsigned int*)(sU.act + row * 256 + kk * 32);
        v8u au;
        #pragma unroll
        for (int rr = 0; rr < 8; ++rr) au[rr] = rowU[afrag_u32(rr, kh)];
        v16bf a = __builtin_bit_cast(v16bf, au);
        const v8u* bp =
            (const v8u*)(W2p + (((size_t)kk * 8 + nt) * 32 + lane) * 16);
        v16bf b = __builtin_bit_cast(v16bf, *bp);
        acc = wmma_bf16(a, b, acc);
      }
      #pragma unroll
      for (int e = 0; e < 8; ++e) {
        float res = sEmb[(mt * 16 + e + 8 * kh) * 128 + col];
        yv[j][e] = acc[e] + res;
      }
    }
  }
  __syncthreads();  // all sU.act reads done; reuse storage for y (f32)
  {
    const int ntbase2 = (wave >> 2) * 4;
    for (int j = 0; j < 4; ++j) {
      int col = (ntbase2 + j) * 16 + m;
      #pragma unroll
      for (int e = 0; e < 8; ++e)
        sU.y[(mt * 16 + e + 8 * kh) * 128 + col] = yv[j][e];
    }
  }
  __syncthreads();

  // Phase 3: LayerNorm, 4 threads per row, cross-lane combine via shfl_xor
  {
    int r = tid >> 2, q = tid & 3;
    const float* yr = sU.y + r * 128 + q * 32;
    float s = 0.f, ss = 0.f;
    #pragma unroll
    for (int c = 0; c < 32; ++c) { float v = yr[c]; s += v; ss += v * v; }
    s += __shfl_xor(s, 1, 32);  ss += __shfl_xor(ss, 1, 32);
    s += __shfl_xor(s, 2, 32);  ss += __shfl_xor(ss, 2, 32);
    float mu = s * (1.f / 128.f);
    float var = ss * (1.f / 128.f) - mu * mu;
    float rstd = rsqrtf(var + 1e-5f);
    float* orow = h_all + (tokbase + r) * H_;
    #pragma unroll
    for (int c = 0; c < 32; ++c) {
      int cc = q * 32 + c;
      orow[cc] = (sU.y[r * 128 + cc] - mu) * rstd * gamma[cc] + beta[cc];
    }
  }
}

// --- Sequential energy-gated delta scan ------------------------------------
// One block per batch, 128 threads; row i of M lives in VGPRs of thread i.
__device__ __forceinline__ float block_sum128(float v, float* red, int tid) {
  #pragma unroll
  for (int w = 16; w > 0; w >>= 1) v += __shfl_xor(v, w, 32);
  if ((tid & 31) == 0) red[tid >> 5] = v;
  __syncthreads();
  float s = red[0] + red[1] + red[2] + red[3];
  __syncthreads();
  return s;
}

__global__ __launch_bounds__(128) void scan_kernel(
    const float* __restrict__ h_all, float* __restrict__ rbuf) {
  __shared__ float sk[128];
  __shared__ float red[4];
  const int b = blockIdx.x, i = threadIdx.x;
  const size_t base = (size_t)b * L_ * H_;
  float Mrow[128];
  #pragma unroll
  for (int j = 0; j < 128; ++j) Mrow[j] = 0.f;

  for (int t = 0; t < L_ - 1; ++t) {
    sk[i] = h_all[base + (size_t)t * H_ + i];
    __syncthreads();
    float vp = 0.f;
    #pragma unroll
    for (int j = 0; j < 128; ++j) vp += Mrow[j] * sk[j];
    float ki = sk[i];
    float kk = block_sum128(ki * ki, red, i);   // ||k||^2 == ||v||^2 (v=k)
    float err = ki - vp / (kk + 1e-6f);
    float es = block_sum128(err * err, red, i);
    if (es > 0.16f * kk) {  // ||err|| > 0.4*||v||, uniform across block
      #pragma unroll
      for (int j = 0; j < 128; ++j) Mrow[j] += err * sk[j];
    }
    __syncthreads();
  }
  sk[i] = h_all[base + (size_t)(L_ - 1) * H_ + i];
  __syncthreads();
  float vp = 0.f;
  #pragma unroll
  for (int j = 0; j < 128; ++j) vp += Mrow[j] * sk[j];
  rbuf[b * H_ + i] = vp;
}

// --- rp = bf16(r @ Wrp + brp), row-major [256][128] ------------------------
__global__ __launch_bounds__(256) void rp_kernel(
    const float* __restrict__ rbuf, const bf16* __restrict__ Wrpp,
    const float* __restrict__ brp, bf16* __restrict__ rp) {
  const int tid = threadIdx.x, wave = tid >> 5, lane = tid & 31;
  const int m = lane & 15, kh = lane >> 4;
  const int mt = blockIdx.x, nt = wave;
  const int row = mt * 16 + m, col = nt * 16 + m;
  float bias = brp[col];
  v8f acc;
  #pragma unroll
  for (int e = 0; e < 8; ++e) acc[e] = bias;
  #pragma unroll
  for (int kk = 0; kk < 4; ++kk) {
    v16bf a;
    #pragma unroll
    for (int rr = 0; rr < 8; ++rr) {
      int kb = kbase(rr, kh);
      a[2 * rr]     = (bf16)rbuf[row * 128 + kk * 32 + kb];
      a[2 * rr + 1] = (bf16)rbuf[row * 128 + kk * 32 + kb + 1];
    }
    const v8u* bp =
        (const v8u*)(Wrpp + (((size_t)kk * 8 + nt) * 32 + lane) * 16);
    v16bf bfr = __builtin_bit_cast(v16bf, *bp);
    acc = wmma_bf16(a, bfr, acc);
  }
  #pragma unroll
  for (int e = 0; e < 8; ++e)
    rp[(mt * 16 + e + 8 * kh) * 128 + col] = (bf16)acc[e];
}

// --- out = rp @ Wout + bout, [256][32000] f32 ------------------------------
__global__ __launch_bounds__(256) void out_kernel(
    const bf16* __restrict__ rp, const bf16* __restrict__ Woutp,
    const float* __restrict__ bout, float* __restrict__ out) {
  const int tid = threadIdx.x, wave = tid >> 5, lane = tid & 31;
  const int m = lane & 15, kh = lane >> 4;
  const int mt = blockIdx.y;
  const int nt = blockIdx.x * 8 + wave;
  const int col = nt * 16 + m;
  const int row = mt * 16 + m;
  float bias = bout[col];
  v8f acc;
  #pragma unroll
  for (int e = 0; e < 8; ++e) acc[e] = bias;
  #pragma unroll
  for (int kk = 0; kk < 4; ++kk) {
    const unsigned int* rowU = (const unsigned int*)(rp + row * 128 + kk * 32);
    v8u au;
    #pragma unroll
    for (int rr = 0; rr < 8; ++rr) au[rr] = rowU[afrag_u32(rr, kh)];
    v16bf a = __builtin_bit_cast(v16bf, au);
    const v8u* bp =
        (const v8u*)(Woutp + (((size_t)kk * 2000 + nt) * 32 + lane) * 16);
    v16bf b = __builtin_bit_cast(v16bf, *bp);
    if (kk < 3)
      __builtin_prefetch(
          Woutp + (((size_t)(kk + 1) * 2000 + nt) * 32 + lane) * 16, 0, 0);
    acc = wmma_bf16(a, b, acc);
  }
  #pragma unroll
  for (int e = 0; e < 8; ++e)
    out[(size_t)(mt * 16 + e + 8 * kh) * V_ + col] = acc[e];
}

extern "C" void kernel_launch(void* const* d_in, const int* in_sizes, int n_in,
                              void* d_out, int out_size, void* d_ws,
                              size_t ws_size, hipStream_t stream) {
  (void)in_sizes; (void)n_in; (void)out_size; (void)ws_size;
  const int*   seq   = (const int*)  d_in[0];
  const float* embed = (const float*)d_in[1];
  const float* W1    = (const float*)d_in[2];
  const float* b1    = (const float*)d_in[3];
  const float* W2    = (const float*)d_in[4];
  const float* b2    = (const float*)d_in[5];
  const float* gamma = (const float*)d_in[6];
  const float* beta  = (const float*)d_in[7];
  const float* Wrp   = (const float*)d_in[8];
  const float* brp   = (const float*)d_in[9];
  const float* Wout  = (const float*)d_in[10];
  const float* bout  = (const float*)d_in[11];
  float* out = (float*)d_out;

  char* w = (char*)d_ws;
  size_t off = 0;
  float* h_all = (float*)(w + off); off += (size_t)B_ * L_ * H_ * 4;  // 256 MB
  bf16* W1p    = (bf16*)(w + off);  off += (size_t)H_ * 2 * H_ * 2;
  bf16* W2p    = (bf16*)(w + off);  off += (size_t)2 * H_ * H_ * 2;
  bf16* Wrpp   = (bf16*)(w + off);  off += (size_t)H_ * H_ * 2;
  bf16* Woutp  = (bf16*)(w + off);  off += (size_t)H_ * V_ * 2;       // 8 MB
  float* rbuf  = (float*)(w + off); off += (size_t)B_ * H_ * 4;
  bf16* rp     = (bf16*)(w + off);  off += (size_t)B_ * H_ * 2;

  pack_b<<<(H_ * 2 * H_ + 255) / 256, 256, 0, stream>>>(W1, W1p, H_, 2 * H_);
  pack_b<<<(2 * H_ * H_ + 255) / 256, 256, 0, stream>>>(W2, W2p, 2 * H_, H_);
  pack_b<<<(H_ * H_ + 255) / 256, 256, 0, stream>>>(Wrp, Wrpp, H_, H_);
  pack_b<<<(H_ * V_ + 255) / 256, 256, 0, stream>>>(Wout, Woutp, H_, V_);

  encoder_kernel<<<(B_ * L_) / 64, 256, 0, stream>>>(
      seq, embed, W1p, b1, W2p, b2, gamma, beta, h_all);

  scan_kernel<<<B_, 128, 0, stream>>>(h_all, rbuf);

  rp_kernel<<<16, 256, 0, stream>>>(rbuf, Wrpp, brp, rp);

  out_kernel<<<dim3(V_ / 16 / 8, B_ / 16), 256, 0, stream>>>(rp, Woutp, bout,
                                                             out);
}
